// nrd_trans_stage5_44744969290342
// MI455X (gfx1250) — matrix operand order repaired and checked by reading, not compiled, not run on hardware
//
#include <hip/hip_runtime.h>
#include <hip/hip_bf16.h>

// ---------------------------------------------------------------------------
// Types for CDNA5 WMMA (wave32): 16x16x32 f16 -> f32
// ---------------------------------------------------------------------------
typedef __attribute__((ext_vector_type(16))) _Float16 v16h;
typedef __attribute__((ext_vector_type(2)))  __fp16   v2h;   // cvt_pkrtz result type
typedef __attribute__((ext_vector_type(8)))  float    v8f;

#define BM 128
#define BN 128
#define BK 32
#define LDSS 40   // halves per LDS row (32 + 8 pad -> conflict-free frag reads)

// ---------------------------------------------------------------------------
// Generic batched GEMM: C = act(alpha * (A @ B') * scaleM + bias [+ residual])
// A: MxK row-major (lda). B: KxN row-major (ldb) or, if transB, B' = Bsrc^T
// with Bsrc NxK row-major (ldb). Batch index z -> (b = z/nh, h = z%nh) with
// independent strides so attention heads (strided into the qkv buffer) and
// plain batched GEMMs both map onto the same kernel.
// biasMode: 0 none, 1 per-N column, 2 per-M row.  act: 0 none, 1 relu, 2 gelu.
// ---------------------------------------------------------------------------
__global__ __launch_bounds__(256) void wmma_gemm_kernel(
    const float* __restrict__ A, const float* __restrict__ B,
    const float* __restrict__ bias, const float* __restrict__ scaleM,
    const float* __restrict__ Rsd, float* __restrict__ C,
    int M, int N, int K, int lda, int ldb, int ldc,
    int transB, int biasMode, int act, float alpha,
    long sAb, long sAh, long sBb, long sBh, long sCb, long sCh, int nh)
{
    __shared__ _Float16 sA[BM * LDSS];
    __shared__ _Float16 sB[BN * LDSS];

    const int bz = blockIdx.z;
    const int bb = bz / nh, hh = bz % nh;
    const float* Ab = A + (long)bb * sAb + (long)hh * sAh;
    const float* Bb = B + (long)bb * sBb + (long)hh * sBh;
    float*       Cb = C + (long)bb * sCb + (long)hh * sCh;
    const float* Rb = Rsd ? (Rsd + (long)bb * sCb + (long)hh * sCh) : nullptr;

    const int m0 = blockIdx.y * BM;
    const int n0 = blockIdx.x * BN;

    const int tid   = threadIdx.x;
    const int lane  = tid & 31;
    const int wid   = tid >> 5;       // 8 waves
    const int waveM = wid >> 2;       // 2 along M (64 rows each)
    const int waveN = wid & 3;        // 4 along N (32 cols each)
    const int lh    = lane & 15;
    const int lhi   = lane >> 4;

    // staging decomposition: fixed per-thread row/column, k-loop only steps ptrs
    const int aRow0 = tid >> 4;          // 0..15 (row advances by 16 per it)
    const int aKp   = (tid & 15) << 1;   // K pair 0,2,..,30
    const int bNn   = tid & 127;         // column for K-major B staging
    const int bK0   = (tid >> 7) << 1;   // 0 or 2 (k advances by 4 per it)

    const bool interiorM = (m0 + BM) <= M;
    const bool interiorN = (n0 + BN) <= N;

    v8f acc[4][2];
    for (int i = 0; i < 4; ++i)
        for (int j = 0; j < 2; ++j)
            acc[i][j] = (v8f)0.0f;

    for (int k0 = 0; k0 < K; k0 += BK) {
        const bool fullK = (k0 + BK) <= K;

        // ---- stage A tile (row-major, f32 -> packed f16) ----
        if (interiorM && fullK) {
            const float* ap = Ab + (long)(m0 + aRow0) * lda + (k0 + aKp);
            const long astep = 16L * lda;
            #pragma unroll
            for (int it = 0; it < 8; ++it) {
                float2 v = *(const float2*)(ap + it * astep);
                *(v2h*)&sA[(aRow0 + it * 16) * LDSS + aKp] =
                    __builtin_amdgcn_cvt_pkrtz(v.x, v.y);
            }
        } else {
            #pragma unroll
            for (int it = 0; it < 8; ++it) {
                int row = aRow0 + it * 16;
                int gr = m0 + row, gk = k0 + aKp;
                int cr  = gr < M ? gr : M - 1;
                int ck0 = gk < K ? gk : K - 1;
                int ck1 = gk + 1 < K ? gk + 1 : K - 1;
                float a0 = Ab[(long)cr * lda + ck0];
                float a1 = Ab[(long)cr * lda + ck1];
                a0 = (gr < M && gk < K)     ? a0 : 0.0f;
                a1 = (gr < M && gk + 1 < K) ? a1 : 0.0f;
                *(v2h*)&sA[row * LDSS + aKp] = __builtin_amdgcn_cvt_pkrtz(a0, a1);
            }
        }

        // ---- stage B tile (stored N-major in LDS) ----
        if (transB) {
            if (interiorN && fullK) {
                const float* bp = Bb + (long)(n0 + aRow0) * ldb + (k0 + aKp);
                const long bstep = 16L * ldb;
                #pragma unroll
                for (int it = 0; it < 8; ++it) {
                    float2 v = *(const float2*)(bp + it * bstep);
                    *(v2h*)&sB[(aRow0 + it * 16) * LDSS + aKp] =
                        __builtin_amdgcn_cvt_pkrtz(v.x, v.y);
                }
            } else {
                #pragma unroll
                for (int it = 0; it < 8; ++it) {
                    int row = aRow0 + it * 16;
                    int gn = n0 + row, gk = k0 + aKp;
                    int cn  = gn < N ? gn : N - 1;
                    int ck0 = gk < K ? gk : K - 1;
                    int ck1 = gk + 1 < K ? gk + 1 : K - 1;
                    float b0 = Bb[(long)cn * ldb + ck0];
                    float b1 = Bb[(long)cn * ldb + ck1];
                    b0 = (gn < N && gk < K)     ? b0 : 0.0f;
                    b1 = (gn < N && gk + 1 < K) ? b1 : 0.0f;
                    *(v2h*)&sB[row * LDSS + aKp] = __builtin_amdgcn_cvt_pkrtz(b0, b1);
                }
            }
        } else {
            if (interiorN && fullK) {
                const float* bp = Bb + (long)(k0 + bK0) * ldb + (n0 + bNn);
                #pragma unroll
                for (int it = 0; it < 8; ++it) {
                    const float* q = bp + (long)(it * 4) * ldb;
                    float b0 = q[0];
                    float b1 = q[ldb];
                    *(v2h*)&sB[bNn * LDSS + bK0 + it * 4] =
                        __builtin_amdgcn_cvt_pkrtz(b0, b1);
                }
            } else {
                #pragma unroll
                for (int it = 0; it < 8; ++it) {
                    int kk = bK0 + it * 4;
                    int gk = k0 + kk, gn = n0 + bNn;
                    int cn  = gn < N ? gn : N - 1;
                    int ck0 = gk < K ? gk : K - 1;
                    int ck1 = gk + 1 < K ? gk + 1 : K - 1;
                    float b0 = Bb[(long)ck0 * ldb + cn];
                    float b1 = Bb[(long)ck1 * ldb + cn];
                    b0 = (gk < K && gn < N)     ? b0 : 0.0f;
                    b1 = (gk + 1 < K && gn < N) ? b1 : 0.0f;
                    *(v2h*)&sB[bNn * LDSS + kk] = __builtin_amdgcn_cvt_pkrtz(b0, b1);
                }
            }
        }

        // speculative prefetch of next K tile (global_prefetch_b8)
        if (k0 + BK < K) {
            __builtin_prefetch(Ab + (long)(m0 + (tid & 127)) * lda + (k0 + BK), 0, 1);
            if (transB)
                __builtin_prefetch(Bb + (long)(n0 + (tid & 127)) * ldb + (k0 + BK), 0, 1);
            else
                __builtin_prefetch(Bb + (long)(k0 + BK + (tid & 31)) * ldb + n0, 0, 1);
        }
        __syncthreads();

        // ---- fragments per the 16-bit 16x32 A / 32x16 B VGPR layouts ----
        union { v16h v; unsigned int u[8]; } fa[4], fb[2];
        #pragma unroll
        for (int mt = 0; mt < 4; ++mt) {
            int row = waveM * 64 + mt * 16 + lh;
            const _Float16* sp = &sA[row * LDSS];
            #pragma unroll
            for (int v = 0; v < 8; ++v) {
                int k = (v < 4 ? 2 * v : 2 * v + 8) + 8 * lhi;
                fa[mt].u[v] = *(const unsigned int*)(sp + k);
            }
        }
        #pragma unroll
        for (int nt = 0; nt < 2; ++nt) {
            int row = waveN * 32 + nt * 16 + lh;
            const _Float16* sp = &sB[row * LDSS];
            #pragma unroll
            for (int v = 0; v < 8; ++v) {
                int k = (v < 4 ? 2 * v : 2 * v + 8) + 8 * lhi;
                fb[nt].u[v] = *(const unsigned int*)(sp + k);
            }
        }
        #pragma unroll
        for (int mt = 0; mt < 4; ++mt)
            #pragma unroll
            for (int nt = 0; nt < 2; ++nt)
                acc[mt][nt] = __builtin_amdgcn_wmma_f32_16x16x32_f16(
                    false, fa[mt].v, false, fb[nt].v,
                    (short)0, acc[mt][nt], false, false);
        __syncthreads();
    }

    // ---- epilogue: C/D layout -> lane lh is N, VGPR r + 8*lhi is M ----
    for (int mt = 0; mt < 4; ++mt) {
        for (int nt = 0; nt < 2; ++nt) {
            int gn = n0 + waveN * 32 + nt * 16 + lh;
            if (gn >= N) continue;
            #pragma unroll
            for (int r = 0; r < 8; ++r) {
                int gm = m0 + waveM * 64 + mt * 16 + r + 8 * lhi;
                if (gm >= M) continue;
                float v = acc[mt][nt][r] * alpha;
                if (scaleM) v *= scaleM[gm];
                if (biasMode == 1) v += bias[gn];
                else if (biasMode == 2) v += bias[gm];
                if (Rb) v += Rb[(long)gm * ldc + gn];
                if (act == 1) v = fmaxf(v, 0.0f);
                else if (act == 2) v = 0.5f * v * (1.0f + erff(v * 0.70710678f));
                Cb[(long)gm * ldc + gn] = v;
            }
        }
    }
}

// ---------------------------------------------------------------------------
// LayerNorm over last dim = 512, one block per row.
// ---------------------------------------------------------------------------
__global__ __launch_bounds__(256) void ln512_kernel(
    const float* __restrict__ x, const float* __restrict__ g,
    const float* __restrict__ b, float* __restrict__ y)
{
    const int row = blockIdx.x;
    const int t = threadIdx.x;
    const float* xr = x + (long)row * 512;
    float v0 = xr[t], v1 = xr[t + 256];
    __shared__ float sred[256];
    sred[t] = v0 + v1; __syncthreads();
    for (int o = 128; o > 0; o >>= 1) { if (t < o) sred[t] += sred[t + o]; __syncthreads(); }
    float mu = sred[0] * (1.0f / 512.0f);
    __syncthreads();
    float d0 = v0 - mu, d1 = v1 - mu;
    sred[t] = d0 * d0 + d1 * d1; __syncthreads();
    for (int o = 128; o > 0; o >>= 1) { if (t < o) sred[t] += sred[t + o]; __syncthreads(); }
    float rs = rsqrtf(sred[0] * (1.0f / 512.0f) + 1e-5f);
    float* yr = y + (long)row * 512;
    yr[t]       = d0 * rs * g[t]       + b[t];
    yr[t + 256] = d1 * rs * g[t + 256] + b[t + 256];
}

// ---------------------------------------------------------------------------
// Row softmax, width = 1024, one block per row (in place).
// ---------------------------------------------------------------------------
__global__ __launch_bounds__(256) void softmax1024_kernel(float* __restrict__ S)
{
    float* r = S + (long)blockIdx.x * 1024;
    const int t = threadIdx.x;
    float v[4]; float m = -1e30f;
    for (int i = 0; i < 4; ++i) { v[i] = r[t + i * 256]; m = fmaxf(m, v[i]); }
    __shared__ float sred[256];
    sred[t] = m; __syncthreads();
    for (int o = 128; o > 0; o >>= 1) { if (t < o) sred[t] = fmaxf(sred[t], sred[t + o]); __syncthreads(); }
    m = sred[0]; __syncthreads();
    float s = 0.0f;
    for (int i = 0; i < 4; ++i) { v[i] = __expf(v[i] - m); s += v[i]; }
    sred[t] = s; __syncthreads();
    for (int o = 128; o > 0; o >>= 1) { if (t < o) sred[t] += sred[t + o]; __syncthreads(); }
    float inv = 1.0f / sred[0];
    for (int i = 0; i < 4; ++i) r[t + i * 256] = v[i] * inv;
}

// ---------------------------------------------------------------------------
// Transposes between NCHW and token layouts.
// ---------------------------------------------------------------------------
__global__ void nchw_to_tokens_kernel(const float* __restrict__ x, float* __restrict__ t,
                                      int B, int C, int P)
{
    long idx = (long)blockIdx.x * 256 + threadIdx.x;
    long total = (long)B * C * P;
    if (idx >= total) return;
    int c = (int)(idx % C); long r = idx / C;
    int p = (int)(r % P);   int b = (int)(r / P);
    t[idx] = x[((long)(b * C + c)) * P + p];
}

__global__ void tokens_to_nchw_kernel(const float* __restrict__ h, float* __restrict__ f,
                                      int B, int C, int P)
{
    long idx = (long)blockIdx.x * 256 + threadIdx.x;
    long total = (long)B * C * P;
    if (idx >= total) return;
    int p = (int)(idx % P); long r = idx / P;
    int c = (int)(r % C);   int b = (int)(r / C);
    f[idx] = h[((long)(b * P + p)) * C + c];
}

// ---------------------------------------------------------------------------
// im2col for 3x3 pad-1 conv: col[(b, c*9 + t9, p)] = in[b, c, y+dy, x+dx]
// ---------------------------------------------------------------------------
__global__ void im2col3x3_kernel(const float* __restrict__ in, float* __restrict__ col,
                                 int B, int C, int H, int W)
{
    long idx = (long)blockIdx.x * 256 + threadIdx.x;
    long hw = (long)H * W;
    long total = (long)B * C * 9 * hw;
    if (idx >= total) return;
    int p  = (int)(idx % hw); long r = idx / hw;
    int t9 = (int)(r % 9);    r /= 9;
    int c  = (int)(r % C);    int b = (int)(r / C);
    int y = p / W, x = p % W;
    int yy = y + t9 / 3 - 1, xx = x + t9 % 3 - 1;
    float v = 0.0f;
    if (yy >= 0 && yy < H && xx >= 0 && xx < W)
        v = in[((long)(b * C + c) * H + yy) * W + xx];
    col[idx] = v;
}

// ---------------------------------------------------------------------------
// Direct 3x3 conv (pad 1) + BN(eps=1e-5, identity stats) + ReLU
// ---------------------------------------------------------------------------
#define BN_RS 0.99999500003750f   // rsqrt(1 + 1e-5)

__global__ void conv3x3_bn_relu_kernel(
    const float* __restrict__ in, const float* __restrict__ w,
    const float* __restrict__ g, const float* __restrict__ bb,
    float* __restrict__ out, int B, int Cin, int Cout, int H, int W)
{
    long idx = (long)blockIdx.x * 256 + threadIdx.x;
    long total = (long)B * Cout * H * W;
    if (idx >= total) return;
    int xx0 = (int)(idx % W); long r = idx / W;
    int yy0 = (int)(r % H);   r /= H;
    int co  = (int)(r % Cout); int b = (int)(r / Cout);
    float s = 0.0f;
    const float* wp = w + (long)co * Cin * 9;
    for (int ci = 0; ci < Cin; ++ci) {
        const float* ip = in + ((long)(b * Cin + ci)) * H * W;
        for (int dy = -1; dy <= 1; ++dy) {
            int yy = yy0 + dy; if (yy < 0 || yy >= H) continue;
            for (int dx = -1; dx <= 1; ++dx) {
                int xx = xx0 + dx; if (xx < 0 || xx >= W) continue;
                s += ip[(long)yy * W + xx] * wp[ci * 9 + (dy + 1) * 3 + (dx + 1)];
            }
        }
    }
    s = s * g[co] * BN_RS + bb[co];
    out[idx] = fmaxf(s, 0.0f);
}

__global__ void dwconv3x3_bn_relu_kernel(
    const float* __restrict__ in, const float* __restrict__ w,
    const float* __restrict__ g, const float* __restrict__ bb,
    float* __restrict__ out, int B, int C, int H, int W)
{
    long idx = (long)blockIdx.x * 256 + threadIdx.x;
    long total = (long)B * C * H * W;
    if (idx >= total) return;
    int xx0 = (int)(idx % W); long r = idx / W;
    int yy0 = (int)(r % H);   r /= H;
    int c   = (int)(r % C);   int b = (int)(r / C);
    const float* ip = in + ((long)(b * C + c)) * H * W;
    const float* wp = w + (long)c * 9;
    float s = 0.0f;
    for (int dy = -1; dy <= 1; ++dy) {
        int yy = yy0 + dy; if (yy < 0 || yy >= H) continue;
        for (int dx = -1; dx <= 1; ++dx) {
            int xx = xx0 + dx; if (xx < 0 || xx >= W) continue;
            s += ip[(long)yy * W + xx] * wp[(dy + 1) * 3 + (dx + 1)];
        }
    }
    s = s * g[c] * BN_RS + bb[c];
    out[idx] = fmaxf(s, 0.0f);
}

// ---------------------------------------------------------------------------
// Dynamic per-pixel MLP head: one workgroup per coarse cell (b, h, w).
// Weights (3126 ch) + the 8x8 x 18ch input patch staged in LDS.
// Layers: 18 -> 16 relu -> 16 relu -> 150.
// ---------------------------------------------------------------------------
__global__ __launch_bounds__(256) void dyn_mlp_kernel(
    const float* __restrict__ abuf, const float* __restrict__ c16,
    const float* __restrict__ catg, const float* __restrict__ catb,
    float* __restrict__ out)
{
    const int blk = blockIdx.x;       // 0..2047
    const int b = blk >> 10;
    const int p = blk & 1023;
    const int hh = p >> 5, ww = p & 31;
    const int t = threadIdx.x;

    __shared__ float w0s[288], b0s[16], w1s[256], b1s[16], w2s[2400], b2s[150];
    __shared__ float xins[64 * 18], h0s[1024], h1s[1024];

    const float* ab = abuf + ((long)b * 3126) * 1024 + p;
    for (int ch = t; ch < 3126; ch += 256) {
        float v = ab[(long)ch * 1024];
        if (ch < 288)       w0s[ch] = v;
        else if (ch < 304)  b0s[ch - 288] = v;
        else if (ch < 560)  w1s[ch - 304] = v;
        else if (ch < 576)  b1s[ch - 560] = v;
        else if (ch < 2976) w2s[ch - 576] = v;
        else                b2s[ch - 2976] = v;
    }
    for (int idx = t; idx < 64 * 18; idx += 256) {
        int sub = idx / 18, ch = idx % 18;
        int i = sub >> 3, j = sub & 7;
        float raw;
        if (ch == 0)      raw = j * 0.125f;
        else if (ch == 1) raw = i * 0.125f;
        else raw = c16[((long)(b * 16 + (ch - 2))) * 65536 + (long)(hh * 8 + i) * 256 + (ww * 8 + j)];
        xins[idx] = raw * catg[ch] * BN_RS + catb[ch];
    }
    __syncthreads();
    for (int idx = t; idx < 1024; idx += 256) {
        int sub = idx >> 4, o = idx & 15;
        float s = b0s[o];
        for (int c = 0; c < 18; ++c) s += xins[sub * 18 + c] * w0s[o * 18 + c];
        h0s[idx] = fmaxf(s, 0.0f);
    }
    __syncthreads();
    for (int idx = t; idx < 1024; idx += 256) {
        int sub = idx >> 4, o = idx & 15;
        float s = b1s[o];
        for (int c = 0; c < 16; ++c) s += h0s[sub * 16 + c] * w1s[o * 16 + c];
        h1s[idx] = fmaxf(s, 0.0f);
    }
    __syncthreads();
    for (int idx = t; idx < 64 * 150; idx += 256) {
        int sub = idx / 150, o = idx % 150;
        int i = sub >> 3, j = sub & 7;
        float s = b2s[o];
        for (int c = 0; c < 16; ++c) s += h1s[sub * 16 + c] * w2s[o * 16 + c];
        out[((long)(b * 150 + o)) * 65536 + (long)(hh * 8 + i) * 256 + (ww * 8 + j)] = s;
    }
}

// ---------------------------------------------------------------------------
// Host-side GEMM launcher
// ---------------------------------------------------------------------------
static inline void launch_gemm(hipStream_t st,
    const float* A, const float* B, const float* bias, const float* scaleM,
    const float* rsd, float* C, int M, int N, int K,
    int lda, int ldb, int ldc, int transB, int biasMode, int act, float alpha,
    long sAb, long sAh, long sBb, long sBh, long sCb, long sCh, int nh, int nbat)
{
    dim3 g((unsigned)((N + BN - 1) / BN), (unsigned)((M + BM - 1) / BM), (unsigned)nbat);
    wmma_gemm_kernel<<<g, 256, 0, st>>>(A, B, bias, scaleM, rsd, C,
        M, N, K, lda, ldb, ldc, transB, biasMode, act, alpha,
        sAb, sAh, sBb, sBh, sCb, sCh, nh);
}

// ---------------------------------------------------------------------------
extern "C" void kernel_launch(void* const* d_in, const int* in_sizes, int n_in,
                              void* d_out, int out_size, void* d_ws, size_t ws_size,
                              hipStream_t stream)
{
    (void)in_sizes; (void)n_in; (void)out_size; (void)ws_size;
    const float* x       = (const float*)d_in[0];
    const float* c1in    = (const float*)d_in[1];
    const float* ln1_g   = (const float*)d_in[2];
    const float* ln1_b   = (const float*)d_in[3];
    const float* qkv_w   = (const float*)d_in[4];
    const float* qkv_b   = (const float*)d_in[5];
    const float* proj_w  = (const float*)d_in[6];
    const float* proj_b  = (const float*)d_in[7];
    const float* ln2_g   = (const float*)d_in[8];
    const float* ln2_b   = (const float*)d_in[9];
    const float* fc1_w   = (const float*)d_in[10];
    const float* fc1_b   = (const float*)d_in[11];
    const float* fc2_w   = (const float*)d_in[12];
    const float* fc2_b   = (const float*)d_in[13];
    const float* norm4_g = (const float*)d_in[14];
    const float* norm4_b = (const float*)d_in[15];
    const float* c1_w1   = (const float*)d_in[16];
    const float* c1_bng  = (const float*)d_in[17];
    const float* c1_bnb  = (const float*)d_in[18];
    const float* c1_w2   = (const float*)d_in[19];
    const float* c1_b2   = (const float*)d_in[20];
    const float* cls_w1  = (const float*)d_in[21];
    const float* cls_bng = (const float*)d_in[22];
    const float* cls_bnb = (const float*)d_in[23];
    const float* cls_w2  = (const float*)d_in[24];
    const float* cls_b2  = (const float*)d_in[25];
    const float* cat_g   = (const float*)d_in[26];
    const float* cat_b   = (const float*)d_in[27];
    const float* smdw_w  = (const float*)d_in[28];
    const float* smdw_g  = (const float*)d_in[29];
    const float* smdw_b  = (const float*)d_in[30];
    const float* smpw_w  = (const float*)d_in[31];
    const float* smpw_g  = (const float*)d_in[32];
    const float* smpw_b  = (const float*)d_in[33];
    float* out = (float*)d_out;

    float* wp = (float*)d_ws;
    auto alloc = [&](long n) { float* r = wp; wp += n; return r; };

    const long TOK = 2048L * 512;
    float* tbuf   = alloc(TOK);              // residual stream (2048, 512)
    float* hln    = alloc(TOK);              // LN output
    float* qkvbuf = alloc(2048L * 1536);     // (2048, 1536)
    float* Sbuf   = alloc(16L * 1024 * 1024);// attention scores (16, 1024, 1024)
    float* attO   = alloc(TOK);              // attention output (2048, 512)
    float* mbuf   = alloc(2048L * 2048);     // MLP hidden
    float* feat   = alloc(TOK);              // (2, 512, 1024) NCHW
    float* clsf   = alloc(TOK);              // cls conv1 output (2, 512, 1024)
    float* colbuf = alloc(2L * 4608 * 1024); // im2col for cls conv
    float* abuf   = alloc(2L * 3126 * 1024); // dynamic weights (2, 3126, 1024)
    float* c48    = alloc(2L * 48 * 65536);  // c1 conv1 output
    float* c16buf = alloc(2L * 16 * 65536);  // c1 conv2 output
    float* segA   = alloc(2L * 150 * 65536); // seg ping
    float* segB   = alloc(2L * 150 * 65536); // seg pong

    // tokens: (2,512,32,32) -> (2048, 512)
    nchw_to_tokens_kernel<<<(unsigned)((2L * 512 * 1024 + 255) / 256), 256, 0, stream>>>(
        x, tbuf, 2, 512, 1024);

    for (int i = 0; i < 6; ++i) {
        // LN1
        ln512_kernel<<<2048, 256, 0, stream>>>(tbuf, ln1_g + i * 512, ln1_b + i * 512, hln);
        // qkv projection
        launch_gemm(stream, hln, qkv_w + (long)i * 512 * 1536, qkv_b + i * 1536,
                    nullptr, nullptr, qkvbuf, 2048, 1536, 512, 512, 1536, 1536,
                    0, 1, 0, 1.0f, 0, 0, 0, 0, 0, 0, 1, 1);
        // S = Q @ K^T * 1/sqrt(64)   (16 = B*heads batched, strided into qkv)
        launch_gemm(stream, qkvbuf, qkvbuf + 512, nullptr, nullptr, nullptr, Sbuf,
                    1024, 1024, 64, 1536, 1536, 1024, 1, 0, 0, 0.125f,
                    1024L * 1536, 64, 1024L * 1536, 64,
                    8L * 1024 * 1024, 1024L * 1024, 8, 16);
        softmax1024_kernel<<<16 * 1024, 256, 0, stream>>>(Sbuf);
        // O = P @ V, heads written into column slices of (2048, 512)
        launch_gemm(stream, Sbuf, qkvbuf + 1024, nullptr, nullptr, nullptr, attO,
                    1024, 64, 1024, 1024, 1536, 512, 0, 0, 0, 1.0f,
                    8L * 1024 * 1024, 1024L * 1024, 1024L * 1536, 64,
                    1024L * 512, 64, 8, 16);
        // proj + residual
        launch_gemm(stream, attO, proj_w + (long)i * 512 * 512, proj_b + i * 512,
                    nullptr, tbuf, tbuf, 2048, 512, 512, 512, 512, 512,
                    0, 1, 0, 1.0f, 0, 0, 0, 0, 0, 0, 1, 1);
        // LN2
        ln512_kernel<<<2048, 256, 0, stream>>>(tbuf, ln2_g + i * 512, ln2_b + i * 512, hln);
        // fc1 + GELU
        launch_gemm(stream, hln, fc1_w + (long)i * 512 * 2048, fc1_b + i * 2048,
                    nullptr, nullptr, mbuf, 2048, 2048, 512, 512, 2048, 2048,
                    0, 1, 2, 1.0f, 0, 0, 0, 0, 0, 0, 1, 1);
        // fc2 + residual
        launch_gemm(stream, mbuf, fc2_w + (long)i * 2048 * 512, fc2_b + i * 512,
                    nullptr, tbuf, tbuf, 2048, 512, 2048, 2048, 512, 512,
                    0, 1, 0, 1.0f, 0, 0, 0, 0, 0, 0, 1, 1);
    }

    // final LN + back to NCHW
    ln512_kernel<<<2048, 256, 0, stream>>>(tbuf, norm4_g, norm4_b, hln);
    tokens_to_nchw_kernel<<<(unsigned)((2L * 512 * 1024 + 255) / 256), 256, 0, stream>>>(
        hln, feat, 2, 512, 1024);

    // cls head: 3x3 conv 512->512 via im2col + WMMA GEMM (BN+ReLU fused),
    // then 1x1 conv 512->3126 as GEMM
    im2col3x3_kernel<<<(unsigned)((2L * 4608 * 1024 + 255) / 256), 256, 0, stream>>>(
        feat, colbuf, 2, 512, 32, 32);
    launch_gemm(stream, cls_w1, colbuf, cls_bnb, cls_bng, nullptr, clsf,
                512, 1024, 4608, 4608, 1024, 1024, 0, 2, 1, BN_RS,
                0, 0, 4608L * 1024, 0, 512L * 1024, 0, 1, 2);
    launch_gemm(stream, cls_w2, clsf, cls_b2, nullptr, nullptr, abuf,
                3126, 1024, 512, 512, 1024, 1024, 0, 2, 0, 1.0f,
                0, 0, 512L * 1024, 0, 3126L * 1024, 0, 1, 2);

    // c1 path: 3x3 conv 64->48 + BN + ReLU (direct), then 1x1 conv 48->16 (GEMM)
    conv3x3_bn_relu_kernel<<<(unsigned)((2L * 48 * 65536 + 255) / 256), 256, 0, stream>>>(
        c1in, c1_w1, c1_bng, c1_bnb, c48, 2, 64, 48, 256, 256);
    launch_gemm(stream, c1_w2, c48, c1_b2, nullptr, nullptr, c16buf,
                16, 65536, 48, 48, 65536, 65536, 0, 2, 0, 1.0f,
                0, 0, 48L * 65536, 0, 16L * 65536, 0, 1, 2);

    // dynamic per-pixel MLP -> segA (2, 150, 256, 256)
    dyn_mlp_kernel<<<2048, 256, 0, stream>>>(abuf, c16buf, cat_g, cat_b, segA);

    // smoothing: 2x (depthwise 3x3 + BN + ReLU, pointwise 150x150 + BN + ReLU)
    for (int i = 0; i < 2; ++i) {
        dwconv3x3_bn_relu_kernel<<<(unsigned)((2L * 150 * 65536 + 255) / 256), 256, 0, stream>>>(
            segA, smdw_w + (long)i * 150 * 9, smdw_g + i * 150, smdw_b + i * 150,
            segB, 2, 150, 256, 256);
        float* dst = (i == 0) ? segA : out;
        // pointwise as GEMM; alpha carries BN rsqrt, scaleM = gamma, bias per-M
        launch_gemm(stream, smpw_w + (long)i * 150 * 150, segB,
                    smpw_b + i * 150, smpw_g + i * 150, nullptr, dst,
                    150, 65536, 150, 150, 65536, 65536, 0, 2, 1, BN_RS,
                    0, 0, 150L * 65536, 0, 150L * 65536, 0, 1, 2);
    }
}
